// QSVSSBlock_73315091743061
// MI455X (gfx1250) — matrix operand order repaired and checked by reading, not compiled, not run on hardware
//
#include <hip/hip_runtime.h>

// ---------------------------------------------------------------------------
// QuadSnake VSS block for MI455X (gfx1250, wave32).
// GEMMs: v_wmma_f32_16x16x32_bf16, f32 accumulate, software-pipelined:
//   A tile  : TENSOR_LOAD_TO_LDS (TDM descriptor DMA, TENSORcnt)
//   B tile  : GLOBAL_LOAD_ASYNC_TO_LDS_B128 (ASYNCcnt)
//   ping-pong LDS buffers, loads for step t+1 issued while computing step t
//   f32 epilogues staged in LDS and written with GLOBAL_STORE_ASYNC_FROM_LDS.
// Activations row-major [m][k] bf16; weights pre-transposed [n][k] bf16 so
// all tile stages are verbatim strided copies (TDM/async friendly).
// ---------------------------------------------------------------------------

#if __has_builtin(__builtin_amdgcn_global_load_async_to_lds_b128)
#define QS_HAVE_ASYNC_LDS 1
#endif
#if __has_builtin(__builtin_amdgcn_global_store_async_from_lds_b128)
#define QS_HAVE_ASYNC_ST 1
#endif
#if __has_builtin(__builtin_amdgcn_s_wait_asynccnt)
#define QS_HAVE_WAIT_ASYNC 1
#endif
#if __has_builtin(__builtin_amdgcn_tensor_load_to_lds) && \
    __has_builtin(__builtin_amdgcn_s_wait_tensorcnt)
#define QS_HAVE_TDM 1
#endif

typedef __bf16 bf16;
typedef bf16  v16bf __attribute__((ext_vector_type(16)));
typedef bf16  v8bf  __attribute__((ext_vector_type(8)));
typedef float v8f   __attribute__((ext_vector_type(8)));
typedef int   v4i   __attribute__((ext_vector_type(4)));
typedef int   v8i   __attribute__((ext_vector_type(8)));
typedef unsigned int v4u __attribute__((ext_vector_type(4)));

#define QS_GPTR(p) ((__attribute__((address_space(1))) v4i*)(p))
#define QS_LPTR(p) ((__attribute__((address_space(3))) v4i*)(p))

#define Bn   8
#define Cc   128
#define Hh   128
#define Ww   128
#define RCc  32
#define HIDc 512
#define Tt   (Hh * Ww)        // 16384
#define Mtot (Bn * Tt)        // 131072
#define KDW  3
#define KSSM 9

#define BM 128
#define BN 64
#define BK 32
#define APITCH 40             // LDS pitch (bf16): 32 data + 8 pad
#define CPITCH 132            // transposed C-stage pitch (f32): 128 + 4 pad

enum { EPI_F32 = 0, EPI_SILU_BF16 = 1, EPI_RES_F32 = 2 };

__device__ __forceinline__ float silu_f(float v) {
    return v / (1.0f + __expf(-v));
}

#if defined(QS_HAVE_ASYNC_LDS)
__device__ __forceinline__ void qs_async_cp16(const void* g, void* l) {
    __builtin_amdgcn_global_load_async_to_lds_b128(QS_GPTR(g), QS_LPTR(l), 0, 0);
}
#endif
#if defined(QS_HAVE_ASYNC_ST)
__device__ __forceinline__ void qs_async_st16(void* g, void* l) {
    __builtin_amdgcn_global_store_async_from_lds_b128(QS_GPTR(g), QS_LPTR(l), 0, 0);
}
#endif

#if defined(QS_HAVE_TDM)
// TDM DMA: 2D tile BM x BK bf16 rows (row stride `rowstride` elems) from
// global into LDS, pad 16dw->+4dw matching APITCH=40 bf16. OOB rows zero.
__device__ __forceinline__ void qs_tdm_load_tile(void* lds, const bf16* gsrc,
                                                 int tdim0, int tdim1,
                                                 int rowstride) {
    unsigned lbase =
        (unsigned)(size_t)(__attribute__((address_space(3))) char*)lds;
    unsigned long long ga = (unsigned long long)(size_t)gsrc;
    v4u g0;
    g0.x = 1u;                                   // count=1, user descriptor
    g0.y = lbase;                                // lds_addr
    g0.z = (unsigned)ga;                         // global_addr[31:0]
    g0.w = (unsigned)(ga >> 32) | (2u << 30);    // global_addr[56:32] | type=2
    v8i g1;
    g1[0] = (1 << 16)      // data_size = 2 bytes
          | (1 << 20)      // pad_enable
          | (3 << 22)      // pad_interval: 16 DWORDs (one 32-bf16 row)
          | (3 << 25);     // pad_amount: 4 DWORDs (8 bf16)
    g1[1] = (tdim0 & 0xFFFF) << 16;                               // dim0 lo
    g1[2] = (int)(((unsigned)tdim0 >> 16) | ((unsigned)(tdim1 & 0xFFFF) << 16));
    g1[3] = (int)(((unsigned)tdim1 >> 16) | ((unsigned)BK << 16)); // tile_dim0
    g1[4] = BM;                                   // tile_dim1 | tile_dim2=0
    g1[5] = rowstride;                            // tensor_dim0_stride lo
    g1[6] = 0;
    g1[7] = 0;
    v4i z4 = {};
#if __clang_major__ >= 23
    v8i z8 = {};
    __builtin_amdgcn_tensor_load_to_lds(g0, g1, z4, z4, z8, 0);
#else
    __builtin_amdgcn_tensor_load_to_lds(g0, g1, z4, z4, 0);
#endif
}
#endif

// ---------------------------------------------------------------------------
// Snake index tables.
// ---------------------------------------------------------------------------
__global__ __launch_bounds__(256)
void qs_build_idx(int* __restrict__ idx) {
    int s = blockIdx.x * blockDim.x + threadIdx.x;
    if (s >= Tt) return;
    int i = s >> 7, j = s & (Ww - 1);
    int col = (i & 1) ? (Ww - 1 - j) : j;
    int p0  = i * Ww + col;
    idx[0 * Tt + s]            = p0;
    idx[1 * Tt + (Tt - 1 - s)] = p0;
    int d    = i + j;
    int imin = d - (Ww - 1); if (imin < 0) imin = 0;
    int imax = (d < Hh) ? d : (Hh - 1);
    int start = 0;
    for (int e = 0; e < d; ++e) {
        int lo = e - (Ww - 1); if (lo < 0) lo = 0;
        int hi = (e < Hh) ? e : (Hh - 1);
        start += hi - lo + 1;
    }
    int q  = (d & 1) ? (imax - i) : (i - imin);
    int s2 = start + q;
    int p  = i * Ww + j;
    idx[2 * Tt + s2]            = p;
    idx[3 * Tt + (Tt - 1 - s2)] = p;
}

// ---------------------------------------------------------------------------
// Weight convert: f32 [K][N] -> bf16 transposed [N][K].
// ---------------------------------------------------------------------------
__global__ __launch_bounds__(256)
void qs_wcvt_t(const float* __restrict__ w, bf16* __restrict__ wt,
               int logK, int N) {
    int i = blockIdx.x * 256 + threadIdx.x;
    int K = 1 << logK;
    if (i >= K * N) return;
    int k = i & (K - 1);
    int n = i >> logK;
    wt[((size_t)n << logK) + k] = (bf16)w[(size_t)k * N + n];
}

// ---------------------------------------------------------------------------
// LN over channels per pixel -> row-major bf16 [b*T+p][c] (16B stores).
// ---------------------------------------------------------------------------
__global__ __launch_bounds__(256)
void qs_ln_channels(const float* __restrict__ x, const float* __restrict__ g,
                    const float* __restrict__ be, bf16* __restrict__ out) {
    int p = blockIdx.x * blockDim.x + threadIdx.x;
    int b = blockIdx.y;
    const float* xb = x + (size_t)b * Cc * Tt + p;
    float s = 0.f, ss = 0.f;
    for (int c = 0; c < Cc; ++c) {
        float v = xb[(size_t)c * Tt];
        s += v; ss += v * v;
    }
    float mean = s * (1.0f / Cc);
    float inv  = rsqrtf(ss * (1.0f / Cc) - mean * mean + 1e-5f);
    bf16* ob = out + ((size_t)b * Tt + p) * Cc;
    for (int c0 = 0; c0 < Cc; c0 += 8) {
        v8bf vv;
#pragma unroll
        for (int j = 0; j < 8; ++j)
            vv[j] = (bf16)((xb[(size_t)(c0 + j) * Tt] - mean) * inv * g[c0 + j] + be[c0 + j]);
        *(v8bf*)(ob + c0) = vv;
    }
}

// ---------------------------------------------------------------------------
// LN over W axis (reference quirk: gamma/beta broadcast along W).
// Block per (b,h): LDS-tiled transpose -> row-major bf16 [b*T+p][c].
// ---------------------------------------------------------------------------
__global__ __launch_bounds__(256)
void qs_ln_rows_t(const float* __restrict__ xr, const float* __restrict__ g,
                  const float* __restrict__ be, bf16* __restrict__ y) {
    __shared__ bf16  tile[Cc][132];
    __shared__ float smean[Cc], sinv[Cc];
    int bh = blockIdx.x;
    int b = bh >> 7, h = bh & (Hh - 1);
    const float* src = xr + (size_t)b * Cc * Tt + (size_t)h * Ww;
    for (int i = threadIdx.x; i < Cc * Ww; i += 256) {
        int w = i & (Ww - 1), c = i >> 7;
        tile[c][w] = (bf16)src[(size_t)c * Tt + w];
    }
    __syncthreads();
    int wave = threadIdx.x >> 5, lane = threadIdx.x & 31;
    for (int c = wave; c < Cc; c += 8) {
        float v0 = (float)tile[c][lane],      v1 = (float)tile[c][lane + 32];
        float v2 = (float)tile[c][lane + 64], v3 = (float)tile[c][lane + 96];
        float s  = v0 + v1 + v2 + v3;
        float ss = v0 * v0 + v1 * v1 + v2 * v2 + v3 * v3;
#pragma unroll
        for (int off = 16; off; off >>= 1) {
            s  += __shfl_xor(s,  off, 32);
            ss += __shfl_xor(ss, off, 32);
        }
        float mean = s * (1.0f / 128.0f);
        if (lane == 0) {
            smean[c] = mean;
            sinv[c]  = rsqrtf(ss * (1.0f / 128.0f) - mean * mean + 1e-5f);
        }
    }
    __syncthreads();
    bf16* dst = y + ((size_t)b * Tt + (size_t)h * Ww) * Cc;
    for (int i = threadIdx.x; i < Cc * Ww; i += 256) {
        int c = i & (Cc - 1), w = i >> 7;
        float v = ((float)tile[c][w] - smean[c]) * sinv[c] * g[w] + be[w];
        dst[(size_t)w * Cc + c] = (bf16)v;
    }
}

// ---------------------------------------------------------------------------
// Depthwise 3x3 + bias + SiLU, channel-major f32.
// ---------------------------------------------------------------------------
__global__ __launch_bounds__(256)
void qs_dwconv_silu(const float* __restrict__ u, const float* __restrict__ wdw,
                    const float* __restrict__ bdw, float* __restrict__ out) {
    int t  = blockIdx.x * blockDim.x + threadIdx.x;
    int ch = blockIdx.y;
    int b  = blockIdx.z;
    int h = t >> 7, w = t & (Ww - 1);
    const float* ub = u + ((size_t)b * RCc + ch) * Tt;
    const float* wk = wdw + ch * (KDW * KDW);
    float a = bdw[ch];
#pragma unroll
    for (int dy = -1; dy <= 1; ++dy)
#pragma unroll
        for (int dx = -1; dx <= 1; ++dx) {
            int hh = h + dy, ww = w + dx;
            if (hh < 0 || hh >= Hh || ww < 0 || ww >= Ww) continue;
            a += wk[(dy + 1) * 3 + (dx + 1)] * ub[hh * Ww + ww];
        }
    out[((size_t)b * RCc + ch) * Tt + t] = silu_f(a);
}

// ---------------------------------------------------------------------------
// Snake SSM: causal conv-9 along permuted sequence, scatter into row-major
// cat buffer [b*T + pixel][4*RC].
// ---------------------------------------------------------------------------
__global__ __launch_bounds__(256)
void qs_ssm_routes(const float* __restrict__ u2, const int* __restrict__ idx,
                   const float* __restrict__ wssm, const float* __restrict__ bssm,
                   bf16* __restrict__ cat) {
    int s  = blockIdx.x * blockDim.x + threadIdx.x;
    int ch = blockIdx.y & (RCc - 1);
    int r  = blockIdx.y >> 5;
    int b  = blockIdx.z;
    const int*   id = idx + (size_t)r * Tt;
    const float* ub = u2 + ((size_t)b * RCc + ch) * Tt;
    const float* wk = wssm + ((size_t)r * RCc + ch) * KSSM;
    float a = bssm[r * RCc + ch];
#pragma unroll
    for (int k = 0; k < KSSM; ++k) {
        int sp = s - (KSSM - 1) + k;
        if (sp >= 0) a += wk[k] * ub[id[sp]];
    }
    cat[((size_t)b * Tt + id[s]) * Cc + r * RCc + ch] = (bf16)a;
}

// ---------------------------------------------------------------------------
// bf16 WMMA GEMM, software-pipelined (ping-pong LDS, one barrier per step).
// A row-major [M][KDIM] bf16 (TDM), Wt transposed [N][KDIM] bf16 (async).
// Block 256 = 8 wave32 (4M x 2N), wave tile 32x32 (2x2 frags), BK=32.
// ---------------------------------------------------------------------------
template <int KDIM, int EPI, bool CHMAJOR_OUT>
__global__ __launch_bounds__(256)
void qs_gemm(const bf16* __restrict__ A, const bf16* __restrict__ Wt,
             const float* __restrict__ bias, const float* __restrict__ res,
             void* __restrict__ outv, int Ntot) {
    __shared__ alignas(16) bf16 As[2][BM][APITCH];
    __shared__ alignas(16) bf16 Bs[2][BN][APITCH];
    __shared__ alignas(16) float Csf[BN * CPITCH];  // fits both epilogue layouts

    const int tid   = threadIdx.x;
    const int wave  = tid >> 5;
    const int lane  = tid & 31;
    const int laneh = lane >> 4;
    const int lanem = lane & 15;
    const int wm    = (wave & 3) * 32;
    const int wn    = (wave >> 2) * 32;

    const int m0 = blockIdx.x * BM;
    const int n0 = blockIdx.y * BN;
    const bf16* Ag = A + (size_t)m0 * KDIM;

    v8f acc[2][2] = {};

    auto load_tiles = [&](int k0, int buf) {
        // ---- A tile: TDM descriptor DMA issued by wave 0 ----
#if defined(QS_HAVE_TDM)
        if (wave == 0)
            qs_tdm_load_tile(&As[buf][0][0], Ag + (size_t)k0, KDIM - k0,
                             Mtot - m0, KDIM);
#elif defined(QS_HAVE_ASYNC_LDS)
        for (int i = tid; i < (BM * BK) / 8; i += 256) {
            int m = i >> 2, kc = (i & 3) * 8;
            qs_async_cp16(Ag + (size_t)m * KDIM + k0 + kc, &As[buf][m][kc]);
        }
#else
        for (int i = tid; i < (BM * BK) / 8; i += 256) {
            int m = i >> 2, kc = (i & 3) * 8;
            *(v8bf*)&As[buf][m][kc] =
                *(const v8bf*)(Ag + (size_t)m * KDIM + k0 + kc);
        }
#endif
        // ---- B tile: async lane copies ----
#if defined(QS_HAVE_ASYNC_LDS)
        for (int i = tid; i < (BN * BK) / 8; i += 256) {
            int n = i >> 2, kc = (i & 3) * 8;
            int ng = n0 + n;
            if (ng < Ntot) {
                qs_async_cp16(Wt + (size_t)ng * KDIM + k0 + kc, &Bs[buf][n][kc]);
            } else {
                v8bf z = {};
                *(v8bf*)&Bs[buf][n][kc] = z;
            }
        }
#else
        for (int i = tid; i < (BN * BK) / 8; i += 256) {
            int n = i >> 2, kc = (i & 3) * 8;
            int ng = n0 + n;
            v8bf v = {};
            if (ng < Ntot) v = *(const v8bf*)(Wt + (size_t)ng * KDIM + k0 + kc);
            *(v8bf*)&Bs[buf][n][kc] = v;
        }
#endif
    };

    constexpr int STEPS = KDIM / BK;
    load_tiles(0, 0);

    for (int t = 0; t < STEPS; ++t) {
        const int buf = t & 1;
        // wait for this buffer's DMA, then rendezvous
#if defined(QS_HAVE_ASYNC_LDS) && defined(QS_HAVE_WAIT_ASYNC)
        __builtin_amdgcn_s_wait_asynccnt(0);
#endif
#if defined(QS_HAVE_TDM)
        __builtin_amdgcn_s_wait_tensorcnt(0);
#endif
        __syncthreads();
        // prefetch next tiles into the other buffer while we compute
        if (t + 1 < STEPS) load_tiles((t + 1) * BK, buf ^ 1);

        v16bf af[2], bfm[2];
#pragma unroll
        for (int mi = 0; mi < 2; ++mi) {
            const bf16* arow = &As[buf][wm + 16 * mi + lanem][0];
            v8bf lo = *(const v8bf*)(arow + 8 * laneh);
            v8bf hi = *(const v8bf*)(arow + 16 + 8 * laneh);
            af[mi] = __builtin_shufflevector(lo, hi, 0, 1, 2, 3, 4, 5, 6, 7,
                                             8, 9, 10, 11, 12, 13, 14, 15);
        }
#pragma unroll
        for (int ni = 0; ni < 2; ++ni) {
            const bf16* brow = &Bs[buf][wn + 16 * ni + lanem][8 * laneh];
            v8bf lo = *(const v8bf*)(brow);
            v8bf hi = *(const v8bf*)(brow + 16);
            bfm[ni] = __builtin_shufflevector(lo, hi, 0, 1, 2, 3, 4, 5, 6, 7,
                                              8, 9, 10, 11, 12, 13, 14, 15);
        }
#pragma unroll
        for (int mi = 0; mi < 2; ++mi)
#pragma unroll
            for (int ni = 0; ni < 2; ++ni)
                acc[mi][ni] = __builtin_amdgcn_wmma_f32_16x16x32_bf16(
                    false, af[mi], false, bfm[ni], (short)0, acc[mi][ni],
                    false, false);
    }

    if (CHMAJOR_OUT) {
        // transposed stage Csf[n][m]: rows = contiguous global m-chunks
#pragma unroll
        for (int mi = 0; mi < 2; ++mi)
#pragma unroll
            for (int ni = 0; ni < 2; ++ni)
#pragma unroll
                for (int i = 0; i < 8; ++i)
                    Csf[(wn + 16 * ni + lanem) * CPITCH +
                        (wm + 16 * mi + 8 * laneh + i)] = acc[mi][ni][i];
        __syncthreads();
        const int b  = m0 / Tt;
        const int p0 = m0 % Tt;
        float* outf = (float*)outv;
        // apply bias (+residual) in LDS
        for (int i = tid; i < BN * BM; i += 256) {
            int m = i & (BM - 1), n = i >> 7;
            int ng = n0 + n;
            if (ng >= Ntot) continue;
            float v = Csf[n * CPITCH + m] + bias[ng];
            size_t off = ((size_t)b * Ntot + ng) * Tt + (size_t)p0 + m;
            if (EPI == EPI_RES_F32) v += res[off];
            Csf[n * CPITCH + m] = v;
        }
        __syncthreads();
#if defined(QS_HAVE_ASYNC_ST)
        // async store from LDS; S_ENDPGM's implicit wait-idle covers completion
        for (int i = tid; i < BN * (BM / 4); i += 256) {
            int n = i >> 5, mc = (i & 31) * 4;
            int ng = n0 + n;
            if (ng >= Ntot) continue;
            qs_async_st16(outf + ((size_t)b * Ntot + ng) * Tt + (size_t)p0 + mc,
                          &Csf[n * CPITCH + mc]);
        }
#else
        for (int i = tid; i < BN * BM; i += 256) {
            int m = i & (BM - 1), n = i >> 7;
            int ng = n0 + n;
            if (ng >= Ntot) continue;
            outf[((size_t)b * Ntot + ng) * Tt + (size_t)p0 + m] =
                Csf[n * CPITCH + m];
        }
#endif
    } else {
        // row-major bf16 out (SiLU), staged Csf[m][n], coalesced along n
#pragma unroll
        for (int mi = 0; mi < 2; ++mi)
#pragma unroll
            for (int ni = 0; ni < 2; ++ni)
#pragma unroll
                for (int i = 0; i < 8; ++i)
                    Csf[(wm + 16 * mi + 8 * laneh + i) * (BN + 1) +
                        (wn + 16 * ni + lanem)] = acc[mi][ni][i];
        __syncthreads();
        for (int i = tid; i < BM * BN; i += 256) {
            int n = i & (BN - 1), m = i >> 6;
            int ng = n0 + n;
            if (ng >= Ntot) continue;
            float v = Csf[m * (BN + 1) + n] + bias[ng];
            if (EPI == EPI_SILU_BF16) v = silu_f(v);
            ((bf16*)outv)[(size_t)(m0 + m) * Ntot + ng] = (bf16)v;
        }
    }
}

// ---------------------------------------------------------------------------
extern "C" void kernel_launch(void* const* d_in, const int* in_sizes, int n_in,
                              void* d_out, int out_size, void* d_ws, size_t ws_size,
                              hipStream_t stream) {
    (void)in_sizes; (void)n_in; (void)out_size; (void)ws_size;
    const float* x       = (const float*)d_in[0];
    const float* gamma1  = (const float*)d_in[1];
    const float* beta1   = (const float*)d_in[2];
    const float* w_proj1 = (const float*)d_in[3];
    const float* b_proj1 = (const float*)d_in[4];
    const float* w_dw    = (const float*)d_in[5];
    const float* b_dw    = (const float*)d_in[6];
    const float* w_ssm   = (const float*)d_in[7];
    const float* b_ssm   = (const float*)d_in[8];
    const float* w_back  = (const float*)d_in[9];
    const float* b_back  = (const float*)d_in[10];
    const float* gamma2  = (const float*)d_in[11];
    const float* beta2   = (const float*)d_in[12];
    const float* w_ffn1  = (const float*)d_in[13];
    const float* b_ffn1  = (const float*)d_in[14];
    const float* w_ffn2  = (const float*)d_in[15];
    const float* b_ffn2  = (const float*)d_in[16];

    char* ws = (char*)d_ws;
    size_t o = 0;
    auto take = [&](size_t bytes) { char* p = ws + o; o += (bytes + 255) & ~(size_t)255; return p; };
    int*  idxbuf  = (int*) take((size_t)4 * Tt * sizeof(int));
    bf16* xln     = (bf16*)take((size_t)Mtot * Cc * sizeof(bf16));   // also y after LN2
    float* u1     = (float*)take((size_t)Mtot * RCc * sizeof(float));
    float* u2     = (float*)take((size_t)Mtot * RCc * sizeof(float));
    bf16* cat     = (bf16*)take((size_t)Mtot * Cc * sizeof(bf16));
    bf16* hmid    = (bf16*)take((size_t)Mtot * HIDc * sizeof(bf16));
    bf16* wtproj  = (bf16*)take((size_t)Cc * RCc * sizeof(bf16));
    bf16* wtback  = (bf16*)take((size_t)Cc * Cc * sizeof(bf16));
    bf16* wtffn1  = (bf16*)take((size_t)Cc * HIDc * sizeof(bf16));
    bf16* wtffn2  = (bf16*)take((size_t)HIDc * Cc * sizeof(bf16));
    float* xres   = (float*)d_out;

    qs_build_idx<<<Tt / 256, 256, 0, stream>>>(idxbuf);

    qs_wcvt_t<<<(Cc * RCc + 255) / 256, 256, 0, stream>>>(w_proj1, wtproj, 7, RCc);
    qs_wcvt_t<<<(Cc * Cc + 255) / 256, 256, 0, stream>>>(w_back, wtback, 7, Cc);
    qs_wcvt_t<<<(Cc * HIDc + 255) / 256, 256, 0, stream>>>(w_ffn1, wtffn1, 7, HIDc);
    qs_wcvt_t<<<(HIDc * Cc + 255) / 256, 256, 0, stream>>>(w_ffn2, wtffn2, 9, Cc);

    qs_ln_channels<<<dim3(Tt / 256, Bn), 256, 0, stream>>>(x, gamma1, beta1, xln);

    // proj1: M=131072, K=128, N=32 -> u1 (channel-major f32)
    qs_gemm<Cc, EPI_F32, true><<<dim3(Mtot / BM, 1), 256, 0, stream>>>(
        xln, wtproj, b_proj1, nullptr, u1, RCc);

    qs_dwconv_silu<<<dim3(Tt / 256, RCc, Bn), 256, 0, stream>>>(u1, w_dw, b_dw, u2);

    qs_ssm_routes<<<dim3(Tt / 256, 4 * RCc, Bn), 256, 0, stream>>>(
        u2, idxbuf, w_ssm, b_ssm, cat);

    // back-proj: K=128, N=128, += x -> x_res (d_out, channel-major f32)
    qs_gemm<Cc, EPI_RES_F32, true><<<dim3(Mtot / BM, Cc / BN), 256, 0, stream>>>(
        cat, wtback, b_back, x, xres, Cc);

    // LN over W axis -> y (row-major bf16, reuses xln)
    qs_ln_rows_t<<<Bn * Hh, 256, 0, stream>>>(xres, gamma2, beta2, xln);

    // FFN1: K=128, N=512, SiLU -> hmid (row-major bf16)
    qs_gemm<Cc, EPI_SILU_BF16, false><<<dim3(Mtot / BM, HIDc / BN), 256, 0, stream>>>(
        xln, wtffn1, b_ffn1, nullptr, hmid, HIDc);

    // FFN2: K=512, N=128, += x_res -> final out (channel-major f32)
    qs_gemm<HIDc, EPI_RES_F32, true><<<dim3(Mtot / BM, Cc / BN), 256, 0, stream>>>(
        hmid, wtffn2, b_ffn2, xres, xres, Cc);
}
